// AttHead_13511967113556
// MI455X (gfx1250) — matrix operand order repaired and checked by reading, not compiled, not run on hardware
//
#include <hip/hip_runtime.h>
#include <hip/hip_bf16.h>
#include <math.h>

typedef _Float16 v16h __attribute__((ext_vector_type(16)));  // 32 B
typedef _Float16 v8h  __attribute__((ext_vector_type(8)));   // 16 B
typedef float    v8f  __attribute__((ext_vector_type(8)));   // WMMA C/D
typedef float    v8fl __attribute__((ext_vector_type(8)));   // 32 B load
typedef float    v4fl __attribute__((ext_vector_type(4)));   // 16 B store

#define EMB   1024
#define HEAD  64
#define SEQ   2048
#define BATCH 8

__device__ inline v8f wmma_f16(v16h a, v16h b, v8f c) {
  return __builtin_amdgcn_wmma_f32_16x16x32_f16(false, a, false, b, (short)0,
                                                c, false, false);
}

// intra-wave LDS store->load ordering fence
__device__ inline void lds_fence() {
  asm volatile("s_wait_dscnt 0x0" ::: "memory");
}

// ---------------------------------------------------------------------------
// A-fragment (16x32 f16) from f32 row-major source: two aligned 32 B runs.
// Lanes 0-15: row M=lane,   elems 0-7 -> K+0..7,  8-15 -> K+16..23
// Lanes16-31: row M=lane-16,elems 0-7 -> K+8..15, 8-15 -> K+24..31
// ---------------------------------------------------------------------------
__device__ inline v16h load_a_f32(const float* __restrict__ src, int row0,
                                  int ld, int k0, int lane) {
  int m  = lane & 15;
  int kq = (lane & 16) ? 8 : 0;
  const float* p = src + (size_t)(row0 + m) * ld + k0 + kq;
  v8fl lo = *(const v8fl*)p;
  v8fl hi = *(const v8fl*)(p + 16);
  v16h a;
#pragma unroll
  for (int i = 0; i < 8; ++i) { a[i] = (_Float16)lo[i]; a[8 + i] = (_Float16)hi[i]; }
  return a;
}

// A-fragment from f16 row-major source: two aligned 16 B runs per lane.
__device__ inline v16h load_a_f16(const _Float16* __restrict__ src, int row0,
                                  int ld, int k0, int lane) {
  int m  = lane & 15;
  int kq = (lane & 16) ? 16 : 0;
  const _Float16* p = src + (size_t)(row0 + m) * ld + k0 + kq;
  v8h lo = *(const v8h*)p;
  v8h hi = *(const v8h*)(p + 8);
  v16h a;
#pragma unroll
  for (int i = 0; i < 8; ++i) { a[i] = lo[i]; a[8 + i] = hi[i]; }
  return a;
}

// ---------------------------------------------------------------------------
// Kernel 0: pack Wq|Wk|Wv (f32) into f16 B-fragment-major layout:
// Wp[frag][lane][i], frag = (e0/32)*12 + w*4 + nt  (12 frags per k-step
// contiguous -> staging reads a contiguous 12 KB region)
// ---------------------------------------------------------------------------
__global__ void pack_weights(const float* __restrict__ Wq,
                             const float* __restrict__ Wk,
                             const float* __restrict__ Wv,
                             _Float16* __restrict__ Wp) {
  int idx = blockIdx.x * blockDim.x + threadIdx.x;
  if (idx >= 3 * EMB * HEAD) return;
  int i    = idx & 15;
  int lane = (idx >> 4) & 31;
  int f    = idx >> 9;
  int nt   = f & 3;
  int w    = (f >> 2) % 3;
  int es   = f / 12;
  int e    = es * 32 + ((lane >> 4) & 1) * 8 + (i & 7) + (i >> 3) * 16;
  int h    = nt * 16 + (lane & 15);
  const float* W = (w == 0) ? Wq : (w == 1) ? Wk : Wv;
  Wp[idx] = (_Float16)W[e * HEAD + h];
}

// ---------------------------------------------------------------------------
// Kernel 1: fused QKV projection.
// 8 waves/block, one 16-row tile per wave. Per-k-step weight fragments
// staged into double-buffered LDS (one barrier per step); B-fragments
// consumed through a rotating-register pipeline so ds_load latency overlaps
// the WMMA chain.
// ---------------------------------------------------------------------------
__global__ void __launch_bounds__(256)
proj_qkv(const float* __restrict__ x, const _Float16* __restrict__ Wp,
         _Float16* __restrict__ qo, _Float16* __restrict__ ko,
         _Float16* __restrict__ vt) {
  __shared__ _Float16 wstage[2 * 12 * 512];  // 24 KB double-buffered
  __shared__ _Float16 ostage[8 * 1024];      // 16 KB per-wave epilogue stage

  const int tid  = threadIdx.x;
  const int lane = tid & 31;
  const int wv   = tid >> 5;
  const int row0 = blockIdx.x * 128 + wv * 16;  // flat row in [BATCH*SEQ]
  const int b    = row0 >> 11;
  const int s0   = row0 & (SEQ - 1);

  v8f acc[12];  // [0..3]=Q, [4..7]=K, [8..11]=V (frag index = w*4+nt)
#pragma unroll
  for (int f = 0; f < 12; ++f) acc[f] = v8f{};

  const int nsteps = EMB / 32;
  const v8h* wsrc = (const v8h*)Wp;  // 12 KB/step = 768 x 16 B units
  v8h ld[3];
#pragma unroll
  for (int u = 0; u < 3; ++u) ld[u] = wsrc[tid + u * 256];

  for (int s = 0; s < nsteps; ++s) {
    _Float16* wbuf = wstage + (s & 1) * (12 * 512);
    v8h* wd = (v8h*)wbuf;
#pragma unroll
    for (int u = 0; u < 3; ++u) wd[tid + u * 256] = ld[u];
    __syncthreads();
    if (s + 1 < nsteps) {
#pragma unroll
      for (int u = 0; u < 3; ++u)
        ld[u] = wsrc[(size_t)(s + 1) * 768 + tid + u * 256];
    }
    v16h a = load_a_f32(x, row0, EMB, s * 32, lane);

    // rotating-register B-fragment pipeline: load f+1 while WMMA f runs
    const _Float16* fb = wbuf + (size_t)lane * 16;  // frag f at fb + f*512
    v16h bf = *(const v16h*)fb;
#pragma unroll
    for (int f = 0; f < 12; ++f) {
      v16h bc = bf;
      if (f + 1 < 12) bf = *(const v16h*)(fb + (size_t)(f + 1) * 512);
      acc[f] = wmma_f16(a, bc, acc[f]);
    }
  }

  const int n    = lane & 15;
  const int half = lane >> 4;
  const int m2   = lane >> 1;
  const int c0   = (lane & 1) * 32;
  _Float16* st   = ostage + wv * 1024;  // per-wave 2 KB (intra-wave reuse)

  // ---- Q ----
#pragma unroll
  for (int nt = 0; nt < 4; ++nt)
#pragma unroll
    for (int r = 0; r < 8; ++r)
      st[(r + 8 * half) * 64 + nt * 16 + n] = (_Float16)acc[nt][r];
  lds_fence();
  *(v16h*)(qo + (size_t)(row0 + m2) * HEAD + c0) = *(const v16h*)(st + m2 * 64 + c0);

  // ---- K ----
#pragma unroll
  for (int nt = 0; nt < 4; ++nt)
#pragma unroll
    for (int r = 0; r < 8; ++r)
      st[(r + 8 * half) * 64 + nt * 16 + n] = (_Float16)acc[4 + nt][r];
  lds_fence();
  *(v16h*)(ko + (size_t)(row0 + m2) * HEAD + c0) = *(const v16h*)(st + m2 * 64 + c0);

  // ---- V transposed: stage as [h][key] (64x16), 32 B per h-row ----
#pragma unroll
  for (int nt = 0; nt < 4; ++nt)
#pragma unroll
    for (int r = 0; r < 8; ++r)
      st[(nt * 16 + n) * 16 + (r + 8 * half)] = (_Float16)acc[8 + nt][r];
  lds_fence();
  _Float16* vtb = vt + (size_t)b * HEAD * SEQ;
#pragma unroll
  for (int j = 0; j < 2; ++j) {
    int h = lane * 2 + j;
    *(v16h*)(vtb + (size_t)h * SEQ + s0) = *(const v16h*)(st + h * 16);
  }
}

// ---------------------------------------------------------------------------
// Kernel 2: causal flash attention.
// 8 waves/block, one 16-query tile per wave; 32-key K/V chunks staged into
// double-buffered LDS (one barrier per chunk), shared by all waves. K frags
// preloaded before score WMMAs; V frag loads issued before the softmax VALU
// section so their LDS latency is hidden.
// ---------------------------------------------------------------------------
__global__ void __launch_bounds__(256)
attn(const _Float16* __restrict__ qg, const _Float16* __restrict__ kg,
     const _Float16* __restrict__ vt, float* __restrict__ out) {
  __shared__ _Float16 kst[2][32 * 64];    // 2 x 4 KB: K chunk [key][h]
  __shared__ _Float16 vst[2][64 * 32];    // 2 x 4 KB: V chunk [h][key]
  __shared__ _Float16 pstg[8][16 * 32];   // 8 KB per-wave P stage
  __shared__ float    ostg[8][16 * 64];   // 32 KB per-wave out stage

  const int tid  = threadIdx.x;
  const int lane = tid & 31;
  const int wv   = tid >> 5;
  const int b    = blockIdx.x >> 4;       // / (SEQ/128)
  const int qblk = blockIdx.x & 15;
  const int q0b  = qblk * 128;
  const int qrow0 = q0b + wv * 16;
  const size_t base   = (size_t)b * SEQ * HEAD;
  const size_t vtbase = (size_t)b * HEAD * SEQ;

  const int n    = lane & 15;
  const int half = lane >> 4;
  const int kq   = half ? 16 : 0;
  const int kb   = lane & 16;

  v16h qa[2];
#pragma unroll
  for (int t = 0; t < 2; ++t)
    qa[t] = load_a_f16(qg + base, qrow0, HEAD, t * 32, lane);

  float mstat[8], lstat[8];
#pragma unroll
  for (int r = 0; r < 8; ++r) { mstat[r] = -__builtin_inff(); lstat[r] = 0.f; }
  v8f oacc[4] = {v8f{}, v8f{}, v8f{}, v8f{}};

  const float scale = 0.03125f;            // EMB^-0.5
  const int nchunk  = (q0b + 128) / 32;    // block-wide causal chunk count
  const int my_kend = qrow0 + 16;

  // staging: each chunk 4 KB K + 4 KB V = 256+256 16 B units (1 each/thread)
  const int vrow = tid >> 2, vun = tid & 3;
  v8h kreg = *(const v8h*)(kg + base + tid * 8);
  v8h vreg = *(const v8h*)(vt + vtbase + (size_t)vrow * SEQ + vun * 8);

  _Float16* pst = pstg[wv];

  for (int c = 0; c < nchunk; ++c) {
    const int key0 = c * 32;
    _Float16* kbuf = kst[c & 1];
    _Float16* vbuf = vst[c & 1];
    ((v8h*)kbuf)[tid] = kreg;
    ((v8h*)vbuf)[tid] = vreg;
    __syncthreads();
    if (c + 1 < nchunk) {
      kreg = *(const v8h*)(kg + base + (size_t)(key0 + 32) * HEAD + tid * 8);
      vreg = *(const v8h*)(vt + vtbase + (size_t)vrow * SEQ + key0 + 32 + vun * 8);
    }

    if (key0 < my_kend) {
      // preload all 4 K^T fragments (distinct regs -> relaxed dscnt waits)
      v16h kf0 = *(const v16h*)(kbuf + (size_t)(n)      * HEAD + kb);
      v16h kf1 = *(const v16h*)(kbuf + (size_t)(16 + n) * HEAD + kb);
      v16h kf2 = *(const v16h*)(kbuf + (size_t)(n)      * HEAD + 32 + kb);
      v16h kf3 = *(const v16h*)(kbuf + (size_t)(16 + n) * HEAD + 32 + kb);
      v8f sc0 = {}, sc1 = {};
      sc0 = wmma_f16(qa[0], kf0, sc0);
      sc1 = wmma_f16(qa[0], kf1, sc1);
      sc0 = wmma_f16(qa[1], kf2, sc0);
      sc1 = wmma_f16(qa[1], kf3, sc1);

      // issue V fragment loads now; latency hidden under the softmax VALU
      v16h vf[4];
#pragma unroll
      for (int nt = 0; nt < 4; ++nt)
        vf[nt] = *(const v16h*)(vbuf + (size_t)(nt * 16 + n) * 32 + kb);

      // online softmax (row stats reduced across each 16-lane half)
#pragma unroll
      for (int r = 0; r < 8; ++r) {
        int m     = r + 8 * half;
        int qglob = qrow0 + m;
        float a0 = (key0 + n      <= qglob) ? sc0[r] * scale : -__builtin_inff();
        float a1 = (key0 + 16 + n <= qglob) ? sc1[r] * scale : -__builtin_inff();
        float vmax = fmaxf(a0, a1);
#pragma unroll
        for (int d = 1; d < 16; d <<= 1)
          vmax = fmaxf(vmax, __shfl_xor(vmax, d, 32));
        float mnew = fmaxf(mstat[r], vmax);
        float p0 = __expf(a0 - mnew);
        float p1 = __expf(a1 - mnew);
        float ps = p0 + p1;
#pragma unroll
        for (int d = 1; d < 16; d <<= 1)
          ps += __shfl_xor(ps, d, 32);
        float corr = __expf(mstat[r] - mnew);
        lstat[r] = lstat[r] * corr + ps;
        mstat[r] = mnew;
#pragma unroll
        for (int nt = 0; nt < 4; ++nt) oacc[nt][r] *= corr;
        pst[m * 32 + n]      = (_Float16)p0;
        pst[m * 32 + 16 + n] = (_Float16)p1;
      }
      lds_fence();

      // reload P in A-fragment layout (two aligned 16 B LDS reads)
      v16h pa;
      {
        const _Float16* p = &pst[n * 32 + kq];
        v8h lo = *(const v8h*)p;
        v8h hi = *(const v8h*)(p + 8);
#pragma unroll
        for (int i = 0; i < 8; ++i) { pa[i] = lo[i]; pa[8 + i] = hi[i]; }
      }

      // out += P[16x32] @ V[32x64]
#pragma unroll
      for (int nt = 0; nt < 4; ++nt)
        oacc[nt] = wmma_f16(pa, vf[nt], oacc[nt]);
    }
  }

  // finalize: 1/l scaling, per-wave LDS stage, coalesced 128 B/lane store
  float* ost = ostg[wv];
#pragma unroll
  for (int r = 0; r < 8; ++r) {
    float inv = 1.f / lstat[r];
#pragma unroll
    for (int nt = 0; nt < 4; ++nt)
      ost[(r + 8 * half) * 64 + nt * 16 + n] = oacc[nt][r] * inv;
  }
  lds_fence();
  {
    int m2 = lane >> 1;
    int c0 = (lane & 1) * 32;
    const float* sp = &ost[m2 * 64 + c0];
    float* gp = out + base + (size_t)(qrow0 + m2) * HEAD + c0;
#pragma unroll
    for (int i = 0; i < 8; ++i) ((v4fl*)gp)[i] = ((const v4fl*)sp)[i];
  }
}

// ---------------------------------------------------------------------------
extern "C" void kernel_launch(void* const* d_in, const int* in_sizes, int n_in,
                              void* d_out, int out_size, void* d_ws,
                              size_t ws_size, hipStream_t stream) {
  const float* x  = (const float*)d_in[0];
  const float* Wq = (const float*)d_in[1];
  const float* Wk = (const float*)d_in[2];
  const float* Wv = (const float*)d_in[3];
  float* out = (float*)d_out;

  const size_t qkv_elems = (size_t)BATCH * SEQ * HEAD;
  _Float16* q  = (_Float16*)d_ws;
  _Float16* k  = q + qkv_elems;
  _Float16* vt = k + qkv_elems;        // transposed V: [B][HEAD][SEQ]
  _Float16* Wp = vt + qkv_elems;       // packed weights: 3*EMB*HEAD f16

  pack_weights<<<(3 * EMB * HEAD + 255) / 256, 256, 0, stream>>>(Wq, Wk, Wv, Wp);
  proj_qkv<<<(BATCH * SEQ) / 128, 256, 0, stream>>>(x, Wp, q, k, vt);
  attn<<<BATCH * (SEQ / 128), 256, 0, stream>>>(q, k, vt, out);
}
